// RSSM_489626272251
// MI455X (gfx1250) — matrix operand order repaired and checked by reading, not compiled, not run on hardware
//
#include <hip/hip_runtime.h>
#include <hip/hip_bf16.h>
#include <math.h>

// ---------------- problem constants ----------------
#define BATCH   64
#define TSTEPS  256
#define DETERD  1024
#define STOCH   32
#define ACTD    64
#define OBSD    1024
#define HIDDEN  512
#define SA_K    96          // STOCH + ACT
#define OUTD    1216        // deter + 6*STOCH

// ---------------- launch configuration ----------------
#define NWG     64
#define TPB     256         // 8 waves per WG -> 2 waves/SIMD with 1 WG/WGP

// ---------------- bf16 weight pool offsets (elements) ----------------
constexpr size_t N_WSA = (size_t)DETERD * SA_K;            // 98304
constexpr size_t N_WIH = (size_t)3 * DETERD * DETERD;      // 3145728
constexpr size_t N_WHH = N_WIH;
constexpr size_t N_WP1 = (size_t)HIDDEN * DETERD;          // 524288
constexpr size_t N_WP2 = (size_t)2 * STOCH * HIDDEN;       // 32768
constexpr size_t N_WQ1 = (size_t)HIDDEN * (DETERD + OBSD); // 1048576
constexpr size_t N_WQ2 = (size_t)2 * STOCH * HIDDEN;       // 32768
constexpr size_t OFF_WSA = 0;
constexpr size_t OFF_WIH = OFF_WSA + N_WSA;
constexpr size_t OFF_WHH = OFF_WIH + N_WIH;
constexpr size_t OFF_WP1 = OFF_WHH + N_WHH;
constexpr size_t OFF_WP2 = OFF_WP1 + N_WP1;
constexpr size_t OFF_WQ1 = OFF_WP2 + N_WP2;
constexpr size_t OFF_WQ2 = OFF_WQ1 + N_WQ1;
constexpr size_t POOL_ELEMS = OFF_WQ2 + N_WQ2;             // 8,028,160

// ---------------- per-WG LDS layout (elements; rows padded +8 for banking) ----------------
constexpr int    S_GRU_LD  = DETERD + 8;                   // 1032
constexpr int    S_SA_LD   = SA_K + 8;                     // 104
constexpr int    S_P1_LD   = DETERD + 8;                   // 1032
constexpr int    S_Q1_LD   = DETERD + OBSD + 8;            // 2056
constexpr size_t S_GRU_OFF = 0;                            // 96 rows (6 gates x 16)
constexpr size_t S_SA_OFF  = S_GRU_OFF + (size_t)96 * S_GRU_LD;   // 99072
constexpr size_t S_H1_OFF  = S_SA_OFF + (size_t)16 * S_SA_LD;     // 100736
constexpr size_t LDS_ELEMS = S_H1_OFF + (size_t)16 * S_Q1_LD;     // 133632
constexpr size_t WEIGHT_LDS_BYTES = LDS_ELEMS * 2;         // 267264
constexpr size_t RED_FLOATS = 4 * 3 * 256;                 // 4 M-tiles x 3 gates x 256
constexpr size_t DYN_LDS_BYTES = WEIGHT_LDS_BYTES + RED_FLOATS * 4; // 279552 < 320KB

// ---------------- WMMA types ----------------
typedef __bf16 bf16_t;
typedef bf16_t v16bf __attribute__((ext_vector_type(16)));
typedef float  v8f   __attribute__((ext_vector_type(8)));

union Frag { v16bf v; uint4 q[2]; };

__device__ inline v8f wmma1(const Frag& a, const Frag& b, v8f c) {
  return __builtin_amdgcn_wmma_f32_16x16x32_bf16(false, a.v, false, b.v,
                                                 (short)0, c, false, false);
}

// Async global->LDS 16B copy (CDNA5 async engine, tracked on ASYNCcnt).
// VDST = per-lane LDS byte address (low 32 bits of generic LDS pointer),
// VADDR = per-lane 64-bit global address, GV mode.
__device__ inline void async_ld16(void* ldsDst, const void* gSrc) {
  const unsigned ldsOff = (unsigned)(unsigned long long)(uintptr_t)ldsDst;
  asm volatile("global_load_async_to_lds_b128 %0, %1, off"
               :: "v"(ldsOff), "v"(gSrc)
               : "memory");
}
__device__ inline void async_wait0() {
  asm volatile("s_wait_asynccnt 0x0" ::: "memory");
}

// Pipelined GEMM tile: A from global (double-buffered, loaded 1 k-tile ahead),
// B operand from an LDS row.  aKOff/bKOff are element offsets of the K origin.
__device__ inline v8f gemm_lds(const __hip_bfloat16* __restrict__ A, int lda,
                               const char* bRow, int aKOff, int bKOff, int kTiles,
                               v8f acc, int lr, int half, int m0)
{
  const char* aRow = (const char*)(A + (size_t)(m0 + lr) * lda);
  Frag a[2], b;
  {
    const int ao = 2 * (aKOff + half * 8);
    a[0].q[0] = *(const uint4*)(aRow + ao);
    a[0].q[1] = *(const uint4*)(aRow + ao + 32);
  }
  for (int kt = 0; kt < kTiles; ++kt) {
    const int cur = kt & 1;
    if (kt + 1 < kTiles) {                       // prefetch next A fragment
      const int ao = 2 * (aKOff + (kt + 1) * 32 + half * 8);
      a[cur ^ 1].q[0] = *(const uint4*)(aRow + ao);
      a[cur ^ 1].q[1] = *(const uint4*)(aRow + ao + 32);
    }
    const int bo = 2 * (bKOff + kt * 32 + half * 8);
    b.q[0] = ((const uint4*)(bRow + bo))[0];
    b.q[1] = ((const uint4*)(bRow + bo))[2];
    acc = wmma1(a[cur], b, acc);
  }
  return acc;
}

// GEMM tile with both operands from global (tiny head GEMMs only).
__device__ inline v8f wmma_acc_g(const __hip_bfloat16* __restrict__ A, int lda,
                                 const __hip_bfloat16* __restrict__ Bw, int ldb,
                                 int m0, int n0, int kTiles, v8f acc, int lane)
{
  const int half = lane >> 4;
  const int lr   = lane & 15;
  const char* aRow = (const char*)(A + (size_t)(m0 + lr) * lda);
  const char* bRow = (const char*)(Bw + (size_t)(n0 + lr) * ldb);
  for (int kt = 0; kt < kTiles; ++kt) {
    const int k0 = kt * 32;
    Frag a, b;
    a.q[0] = *(const uint4*)(aRow + 2 * (k0 + half * 8));
    a.q[1] = *(const uint4*)(aRow + 2 * (k0 + 16 + half * 8));
    const uint4* bp = (const uint4*)(bRow + 2 * (k0 + half * 8));
    b.q[0] = bp[0];
    b.q[1] = bp[2];
    acc = wmma1(a, b, acc);
  }
  return acc;
}

// ---------------- device-wide sense barrier (graph-capture safe) ----------------
__device__ inline void grid_barrier(unsigned* __restrict__ bar)
{
  __syncthreads();
  __threadfence();
  if (threadIdx.x == 0) {
    unsigned gen = __hip_atomic_load(&bar[1], __ATOMIC_ACQUIRE, __HIP_MEMORY_SCOPE_AGENT);
    unsigned arr = __hip_atomic_fetch_add(&bar[0], 1u, __ATOMIC_ACQ_REL, __HIP_MEMORY_SCOPE_AGENT);
    if (arr == NWG - 1) {
      __hip_atomic_store(&bar[0], 0u, __ATOMIC_RELAXED, __HIP_MEMORY_SCOPE_AGENT);
      __hip_atomic_store(&bar[1], gen + 1u, __ATOMIC_RELEASE, __HIP_MEMORY_SCOPE_AGENT);
    } else {
      while (__hip_atomic_load(&bar[1], __ATOMIC_ACQUIRE, __HIP_MEMORY_SCOPE_AGENT) == gen)
        __builtin_amdgcn_s_sleep(1);
    }
  }
  __syncthreads();
  __threadfence();
}

__device__ inline float sigmoidf_(float x) { return 1.0f / (1.0f + expf(-x)); }
__device__ inline float softplusf_(float x) { return (x > 20.0f) ? x : log1pf(expf(x)); }

// ---------------- prep: fp32 -> bf16 weights, zero state, reset barrier ----------------
__global__ void rssm_prep(const float* __restrict__ Wsa, const float* __restrict__ Wih,
                          const float* __restrict__ Whh, const float* __restrict__ Wp1,
                          const float* __restrict__ Wp2, const float* __restrict__ Wq1,
                          const float* __restrict__ Wq2,
                          __hip_bfloat16* __restrict__ pool,
                          float* __restrict__ deterF,
                          __hip_bfloat16* __restrict__ d0, __hip_bfloat16* __restrict__ d1,
                          __hip_bfloat16* __restrict__ sa, unsigned* __restrict__ bar)
{
  size_t gt = (size_t)blockIdx.x * blockDim.x + threadIdx.x;
  size_t stride = (size_t)gridDim.x * blockDim.x;
  for (size_t i = gt; i < N_WSA; i += stride) pool[OFF_WSA + i] = __float2bfloat16(Wsa[i]);
  for (size_t i = gt; i < N_WIH; i += stride) pool[OFF_WIH + i] = __float2bfloat16(Wih[i]);
  for (size_t i = gt; i < N_WHH; i += stride) pool[OFF_WHH + i] = __float2bfloat16(Whh[i]);
  for (size_t i = gt; i < N_WP1; i += stride) pool[OFF_WP1 + i] = __float2bfloat16(Wp1[i]);
  for (size_t i = gt; i < N_WP2; i += stride) pool[OFF_WP2 + i] = __float2bfloat16(Wp2[i]);
  for (size_t i = gt; i < N_WQ1; i += stride) pool[OFF_WQ1 + i] = __float2bfloat16(Wq1[i]);
  for (size_t i = gt; i < N_WQ2; i += stride) pool[OFF_WQ2 + i] = __float2bfloat16(Wq2[i]);
  const __hip_bfloat16 z = __float2bfloat16(0.0f);
  for (size_t i = gt; i < (size_t)BATCH * DETERD; i += stride) {
    deterF[i] = 0.0f; d0[i] = z; d1[i] = z;
  }
  for (size_t i = gt; i < (size_t)BATCH * SA_K; i += stride) sa[i] = z;
  if (gt < 2) bar[gt] = 0u;
}

// ---------------- persistent rollout kernel ----------------
__global__ __launch_bounds__(TPB, 1)
void rssm_rollout(const float* __restrict__ obs,  const float* __restrict__ act,
                  const float* __restrict__ n_p,  const float* __restrict__ n_q,
                  const float* __restrict__ b_sa, const float* __restrict__ b_ih,
                  const float* __restrict__ b_hh, const float* __restrict__ bp1,
                  const float* __restrict__ bp2,  const float* __restrict__ bq1,
                  const float* __restrict__ bq2,
                  const __hip_bfloat16* __restrict__ pool,
                  __hip_bfloat16* __restrict__ sa,   __hip_bfloat16* __restrict__ emb,
                  __hip_bfloat16* __restrict__ d0,   __hip_bfloat16* __restrict__ d1,
                  __hip_bfloat16* __restrict__ obsb,
                  __hip_bfloat16* __restrict__ h1p,  __hip_bfloat16* __restrict__ h1q,
                  float* __restrict__ deterF, float* __restrict__ pmS, float* __restrict__ qmS,
                  unsigned* __restrict__ bar, float* __restrict__ out)
{
  const int tid  = threadIdx.x;
  const int g    = blockIdx.x;             // WG owns one 16-wide output column tile
  const int gt   = g * TPB + tid;
  const int w    = tid >> 5;               // wave in WG (0..7)
  const int lane = tid & 31;
  const int gw   = g * 8 + w;              // global wave id 0..511
  const int lr   = lane & 15;
  const int half = lane >> 4;
  const int mw   = w & 3;                  // M-tile (0..3)
  const bool hi  = w >= 4;                 // hi waves: hidden-gate / second-K half
  const int m0   = mw * 16;

  const __hip_bfloat16* Wsa = pool + OFF_WSA;
  const __hip_bfloat16* Wih = pool + OFF_WIH;
  const __hip_bfloat16* Whh = pool + OFF_WHH;
  const __hip_bfloat16* Wp1 = pool + OFF_WP1;
  const __hip_bfloat16* Wp2 = pool + OFF_WP2;
  const __hip_bfloat16* Wq1 = pool + OFF_WQ1;
  const __hip_bfloat16* Wq2 = pool + OFF_WQ2;

  extern __shared__ __align__(16) char smem[];
  __hip_bfloat16* sGRU = (__hip_bfloat16*)smem;        // 96 rows x S_GRU_LD
  __hip_bfloat16* sSA  = sGRU + S_SA_OFF;              // 16 rows x S_SA_LD
  __hip_bfloat16* sH1  = sGRU + S_H1_OFF;              // 16 rows x (S_P1_LD | S_Q1_LD)
  float*          sRED = (float*)(smem + WEIGHT_LDS_BYTES); // cross-wave exchange

  // ===== one-time weight staging into LDS via async global->LDS engine =====
  for (int idx = tid; idx < 96 * 128; idx += TPB) {    // GRU: 6 gates x 16 rows x 1024
    const int r = idx >> 7, c = idx & 127;
    const int gi = r >> 4, rr = r & 15;
    const __hip_bfloat16* src = (gi < 3)
        ? Wih + ((size_t)gi * DETERD + (size_t)g * 16 + rr) * DETERD
        : Whh + ((size_t)(gi - 3) * DETERD + (size_t)g * 16 + rr) * DETERD;
    async_ld16(sGRU + (size_t)r * S_GRU_LD + c * 8, src + c * 8);
  }
  for (int idx = tid; idx < 16 * 12; idx += TPB) {     // W_sa: 16 rows x 96
    const int r = idx / 12, c = idx % 12;
    async_ld16(sSA + (size_t)r * S_SA_LD + c * 8,
               Wsa + ((size_t)g * 16 + r) * SA_K + c * 8);
  }
  if (g < 32) {                                        // Wp1 slice: 16 x 1024
    for (int idx = tid; idx < 16 * 128; idx += TPB) {
      const int r = idx >> 7, c = idx & 127;
      async_ld16(sH1 + (size_t)r * S_P1_LD + c * 8,
                 Wp1 + ((size_t)g * 16 + r) * DETERD + c * 8);
    }
  } else {                                             // Wq1 slice: 16 x 2048
    for (int idx = tid; idx < 16 * 256; idx += TPB) {
      const int r = idx >> 8, c = idx & 255;
      async_ld16(sH1 + (size_t)r * S_Q1_LD + c * 8,
                 Wq1 + ((size_t)(g - 32) * 16 + r) * (DETERD + OBSD) + c * 8);
    }
  }
  async_wait0();
  __syncthreads();

  // D2 epilogue (softplus / sampling / outputs / next stoch)
  auto do_D2 = [&](int tt) {
    for (int i = gt; i < BATCH * STOCH; i += NWG * TPB) {
      const int b = i >> 5, j = i & 31;
      const size_t ob = ((size_t)b * TSTEPS + tt) * OUTD + DETERD;
      const size_t nb = ((size_t)b * TSTEPS + tt) * STOCH + j;
      const float pm  = pmS[b * 2 * STOCH + j];
      const float ps  = softplusf_(pmS[b * 2 * STOCH + STOCH + j]) + 1e-5f;
      const float pst = pm + ps * n_p[nb];
      out[ob +   0 + j] = pm;
      out[ob +  32 + j] = ps;
      out[ob +  64 + j] = pst;
      const float qm  = qmS[b * 2 * STOCH + j];
      const float qs  = softplusf_(qmS[b * 2 * STOCH + STOCH + j]) + 1e-5f;
      const float qst = qm + qs * n_q[nb];
      out[ob +  96 + j] = qm;
      out[ob + 128 + j] = qs;
      out[ob + 160 + j] = qst;
      sa[b * SA_K + j] = __float2bfloat16(qst);        // stoch carry
    }
  };

  for (int t = 0; t < TSTEPS; ++t) {
    __hip_bfloat16* dcur = (t & 1) ? d1 : d0;
    __hip_bfloat16* dnxt = (t & 1) ? d0 : d1;

    // ---- Phase 0: previous step's epilogue + stage act/obs as bf16 ----
    if (t > 0) do_D2(t - 1);
    for (int i = gt; i < BATCH * ACTD; i += NWG * TPB) {
      const int b = i >> 6, j = i & 63;
      sa[b * SA_K + STOCH + j] =
          __float2bfloat16(act[((size_t)b * TSTEPS + t) * ACTD + j]);
    }
    for (int i = gt; i < BATCH * OBSD; i += NWG * TPB) {
      const int b = i >> 10, j = i & 1023;
      obsb[i] = __float2bfloat16(obs[((size_t)b * TSTEPS + t) * OBSD + j]);
      if (t + 1 < TSTEPS)
        __builtin_prefetch(obs + ((size_t)b * TSTEPS + (t + 1)) * OBSD + j, 0, 1);
    }
    grid_barrier(bar);

    // ---- Phase A: emb = relu([stoch,act] @ Wsa^T + b_sa)  (64x1024, K=96) ----
    if (!hi) {
      const int n0 = g * 16;
      v8f acc = {};
      acc = gemm_lds(sa, SA_K, (const char*)(sSA + (size_t)lr * S_SA_LD),
                     0, 0, SA_K / 32, acc, lr, half, m0);
      const float bias = b_sa[n0 + lr];
      for (int v = 0; v < 8; ++v) {
        float y = acc[v] + bias;
        y = y > 0.0f ? y : 0.0f;
        emb[(size_t)(m0 + half * 8 + v) * DETERD + n0 + lr] = __float2bfloat16(y);
      }
    }
    grid_barrier(bar);

    // ---- Phase B: GRU. lo waves: 3 input gates; hi waves: 3 hidden gates ----
    {
      const int n0 = g * 16;
      v8f acc0 = {}, acc1 = {}, acc2 = {};
      const char* aRow = (const char*)((hi ? dcur : emb) + (size_t)(m0 + lr) * DETERD);
      const int gbase = hi ? 3 : 0;
      const char* l0 = (const char*)(sGRU + (size_t)((gbase + 0) * 16 + lr) * S_GRU_LD);
      const char* l1 = (const char*)(sGRU + (size_t)((gbase + 1) * 16 + lr) * S_GRU_LD);
      const char* l2 = (const char*)(sGRU + (size_t)((gbase + 2) * 16 + lr) * S_GRU_LD);
      Frag a[2], b;
      {
        const int ao = 2 * (half * 8);
        a[0].q[0] = *(const uint4*)(aRow + ao);
        a[0].q[1] = *(const uint4*)(aRow + ao + 32);
      }
      for (int kt = 0; kt < DETERD / 32; ++kt) {
        const int cur = kt & 1;
        if (kt + 1 < DETERD / 32) {              // prefetch next A fragment
          const int ao = 2 * ((kt + 1) * 32 + half * 8);
          a[cur ^ 1].q[0] = *(const uint4*)(aRow + ao);
          a[cur ^ 1].q[1] = *(const uint4*)(aRow + ao + 32);
        }
        const int bo = 2 * (kt * 32 + half * 8);
        b.q[0] = ((const uint4*)(l0 + bo))[0]; b.q[1] = ((const uint4*)(l0 + bo))[2];
        acc0 = wmma1(a[cur], b, acc0);
        b.q[0] = ((const uint4*)(l1 + bo))[0]; b.q[1] = ((const uint4*)(l1 + bo))[2];
        acc1 = wmma1(a[cur], b, acc1);
        b.q[0] = ((const uint4*)(l2 + bo))[0]; b.q[1] = ((const uint4*)(l2 + bo))[2];
        acc2 = wmma1(a[cur], b, acc2);
      }
      float* red = sRED + (size_t)mw * 768;      // 3 gates x 256 floats per M-tile
      if (hi) {
        for (int v = 0; v < 8; ++v) {
          red[0 * 256 + v * 32 + lane] = acc0[v];
          red[1 * 256 + v * 32 + lane] = acc1[v];
          red[2 * 256 + v * 32 + lane] = acc2[v];
        }
      }
      __syncthreads();
      if (!hi) {
        const float bir = b_ih[n0 + lr], biz = b_ih[DETERD + n0 + lr], bin = b_ih[2 * DETERD + n0 + lr];
        const float bhr = b_hh[n0 + lr], bhz = b_hh[DETERD + n0 + lr], bhn = b_hh[2 * DETERD + n0 + lr];
        for (int v = 0; v < 8; ++v) {
          const int m = m0 + half * 8 + v;       // batch row
          const int n = n0 + lr;                 // deter column
          const float hr = red[0 * 256 + v * 32 + lane];
          const float hz = red[1 * 256 + v * 32 + lane];
          const float hn = red[2 * 256 + v * 32 + lane];
          const float r  = sigmoidf_(acc0[v] + bir + hr + bhr);
          const float z  = sigmoidf_(acc1[v] + biz + hz + bhz);
          const float nn = tanhf(acc2[v] + bin + r * (hn + bhn));
          const float h  = deterF[(size_t)m * DETERD + n];
          const float dn = (1.0f - z) * nn + z * h;
          deterF[(size_t)m * DETERD + n] = dn;
          dnxt[(size_t)m * DETERD + n]   = __float2bfloat16(dn);
          out[((size_t)m * TSTEPS + t) * OUTD + n] = dn;
        }
      }
    }
    grid_barrier(bar);

    // ---- Phase C: hidden layers; K split across lo/hi waves, LDS reduce ----
    if (g < 32) {              // h1p = relu(deter @ Wp1^T + bp1)  (64x512, K=1024)
      const int n0 = g * 16;
      const char* bRow = (const char*)(sH1 + (size_t)lr * S_P1_LD);
      const int kOff = hi ? (DETERD / 2) : 0;
      v8f acc = {};
      acc = gemm_lds(dnxt, DETERD, bRow, kOff, kOff, DETERD / 64, acc, lr, half, m0);
      float* red = sRED + (size_t)mw * 256;
      if (hi) {
        for (int v = 0; v < 8; ++v) red[v * 32 + lane] = acc[v];
      }
      __syncthreads();
      if (!hi) {
        const float bias = bp1[n0 + lr];
        for (int v = 0; v < 8; ++v) {
          float y = acc[v] + red[v * 32 + lane] + bias;
          y = y > 0.0f ? y : 0.0f;
          h1p[(size_t)(m0 + half * 8 + v) * HIDDEN + n0 + lr] = __float2bfloat16(y);
        }
      }
    } else {                   // h1q = relu([deter,obs] @ Wq1^T + bq1)  (64x512, K=2048)
      const int n0 = (g - 32) * 16;
      const char* bRow = (const char*)(sH1 + (size_t)lr * S_Q1_LD);
      v8f acc = {};
      if (hi) acc = gemm_lds(obsb, OBSD,   bRow, 0, DETERD, OBSD  / 32, acc, lr, half, m0);
      else    acc = gemm_lds(dnxt, DETERD, bRow, 0, 0,      DETERD / 32, acc, lr, half, m0);
      float* red = sRED + (size_t)mw * 256;
      if (hi) {
        for (int v = 0; v < 8; ++v) red[v * 32 + lane] = acc[v];
      }
      __syncthreads();
      if (!hi) {
        const float bias = bq1[n0 + lr];
        for (int v = 0; v < 8; ++v) {
          float y = acc[v] + red[v * 32 + lane] + bias;
          y = y > 0.0f ? y : 0.0f;
          h1q[(size_t)(m0 + half * 8 + v) * HIDDEN + n0 + lr] = __float2bfloat16(y);
        }
      }
    }
    grid_barrier(bar);

    // ---- Phase D1: head output GEMMs (64x64 each, K=512) on 32 waves ----
    if (gw < 32) {
      const bool isQ = gw >= 16;
      const int  w2  = isQ ? gw - 16 : gw;
      const int  tm0 = (w2 & 3) * 16;
      const int  tn0 = (w2 >> 2) * 16;
      v8f acc = {};
      acc = wmma_acc_g(isQ ? h1q : h1p, HIDDEN, isQ ? Wq2 : Wp2, HIDDEN,
                       tm0, tn0, HIDDEN / 32, acc, lane);
      const float bias = (isQ ? bq2 : bp2)[tn0 + lr];
      float* dst = isQ ? qmS : pmS;
      for (int v = 0; v < 8; ++v)
        dst[(size_t)(tm0 + half * 8 + v) * (2 * STOCH) + tn0 + lr] = acc[v] + bias;
    }
    grid_barrier(bar);
  }

  // final epilogue for t = T-1
  do_D2(TSTEPS - 1);
}

// ---------------- host-side launch ----------------
extern "C" void kernel_launch(void* const* d_in, const int* in_sizes, int n_in,
                              void* d_out, int out_size, void* d_ws, size_t ws_size,
                              hipStream_t stream)
{
  const float* obs  = (const float*)d_in[0];
  const float* act  = (const float*)d_in[1];
  const float* n_p  = (const float*)d_in[2];
  const float* n_q  = (const float*)d_in[3];
  const float* Wsa  = (const float*)d_in[4];
  const float* b_sa = (const float*)d_in[5];
  const float* Wih  = (const float*)d_in[6];
  const float* b_ih = (const float*)d_in[7];
  const float* Whh  = (const float*)d_in[8];
  const float* b_hh = (const float*)d_in[9];
  const float* Wp1  = (const float*)d_in[10];
  const float* bp1  = (const float*)d_in[11];
  const float* Wp2  = (const float*)d_in[12];
  const float* bp2  = (const float*)d_in[13];
  const float* Wq1  = (const float*)d_in[14];
  const float* bq1  = (const float*)d_in[15];
  const float* Wq2  = (const float*)d_in[16];
  const float* bq2  = (const float*)d_in[17];
  float* out = (float*)d_out;

  char* ws = (char*)d_ws;
  size_t off = 0;
  auto carve = [&](size_t bytes) -> void* {
    void* p = (void*)(ws + off);
    off = (off + bytes + 255) & ~(size_t)255;
    return p;
  };
  unsigned*       bar    = (unsigned*)carve(2 * sizeof(unsigned));
  __hip_bfloat16* pool   = (__hip_bfloat16*)carve(POOL_ELEMS * 2);
  __hip_bfloat16* sa     = (__hip_bfloat16*)carve((size_t)BATCH * SA_K * 2);
  __hip_bfloat16* emb    = (__hip_bfloat16*)carve((size_t)BATCH * DETERD * 2);
  __hip_bfloat16* d0     = (__hip_bfloat16*)carve((size_t)BATCH * DETERD * 2);
  __hip_bfloat16* d1     = (__hip_bfloat16*)carve((size_t)BATCH * DETERD * 2);
  __hip_bfloat16* obsb   = (__hip_bfloat16*)carve((size_t)BATCH * OBSD * 2);
  __hip_bfloat16* h1p    = (__hip_bfloat16*)carve((size_t)BATCH * HIDDEN * 2);
  __hip_bfloat16* h1q    = (__hip_bfloat16*)carve((size_t)BATCH * HIDDEN * 2);
  float*          deterF = (float*)carve((size_t)BATCH * DETERD * 4);
  float*          pmS    = (float*)carve((size_t)BATCH * 2 * STOCH * 4);
  float*          qmS    = (float*)carve((size_t)BATCH * 2 * STOCH * 4);
  (void)ws_size; (void)in_sizes; (void)n_in; (void)out_size;

  rssm_prep<<<1024, 256, 0, stream>>>(Wsa, Wih, Whh, Wp1, Wp2, Wq1, Wq2,
                                      pool, deterF, d0, d1, sa, bar);
  rssm_rollout<<<NWG, TPB, DYN_LDS_BYTES, stream>>>(obs, act, n_p, n_q,
                                        b_sa, b_ih, b_hh, bp1, bp2, bq1, bq2,
                                        pool, sa, emb, d0, d1, obsb, h1p, h1q,
                                        deterF, pmS, qmS, bar, out);
}
